// RAttentionLayer_83021717832113
// MI455X (gfx1250) — matrix operand order repaired and checked by reading, not compiled, not run on hardware
//
#include <hip/hip_runtime.h>
#include <hip/hip_bf16.h>

// ---------------------------------------------------------------------------
// RAttention layer for MI455X (gfx1250, wave32, WMMA + async LDS copies).
// B=2, L=512, H=8, D=16, HD=128, HID=128.  SCALE = 0.25.
// ---------------------------------------------------------------------------

#define Bn   2
#define Ln   512
#define Hn   8
#define Dn   16
#define HIDn 128
#define HDn  128
#define NTOK (Bn * Ln)          // 1024
#define EPSc 1e-5f

typedef __attribute__((ext_vector_type(16))) _Float16 v16h;
typedef __attribute__((ext_vector_type(8)))  _Float16 v8h;
typedef __attribute__((ext_vector_type(8)))  float    v8f;

// ---------------------------------------------------------------------------
// CDNA5 async global->LDS copy (ASYNCcnt-tracked, no VGPR round trip).
// Per-lane LDS byte address in a VGPR, per-lane 64-bit global address (GV).
// ---------------------------------------------------------------------------
__device__ __forceinline__ void async_ld_b128(unsigned lds_off, const void* g) {
  asm volatile("global_load_async_to_lds_b128 %0, %1, off"
               :: "v"(lds_off), "v"((unsigned long long)(uintptr_t)g)
               : "memory");
}
__device__ __forceinline__ void wait_async_le4()  {
  asm volatile("s_wait_asynccnt 0x4" ::: "memory");
}
__device__ __forceinline__ void wait_async_le0()  {
  asm volatile("s_wait_asynccnt 0x0" ::: "memory");
}
__device__ __forceinline__ unsigned lds_addr(const void* p) {
  return (unsigned)(uintptr_t)p;    // generic->LDS: hw uses addr[31:0]
}

// ---------------------------------------------------------------------------
// WMMA fragment helpers (CDNA5 16x16x32 f16, wave32 layouts per ISA 7.12.2)
// ---------------------------------------------------------------------------
__device__ __forceinline__ v16h frag_cat(v8h lo, v8h hi) {
  return __builtin_shufflevector(lo, hi, 0, 1, 2, 3, 4, 5, 6, 7,
                                         8, 9, 10, 11, 12, 13, 14, 15);
}

// A fragment (16x32) from f16 LDS, row-major, leading dim lda.
__device__ __forceinline__ v16h load_a_frag(const _Float16* a, int lda, int k0) {
  const int lane = threadIdx.x & 31;
  const int row  = lane & 15;
  const int kh   = lane >> 4;
  const _Float16* p = a + row * lda + k0 + kh * 8;
  v8h lo = *(const v8h*)(p);
  v8h hi = *(const v8h*)(p + 16);
  return frag_cat(lo, hi);
}

// A fragment (16x32) from f32 LDS (async-staged r tile), converting to f16.
__device__ __forceinline__ v16h load_a_frag_f32(const float* a, int lda, int k0) {
  const int lane = threadIdx.x & 31;
  const int row  = lane & 15;
  const int kh   = lane >> 4;
  const float* p = a + row * lda + k0 + kh * 8;
  const float4 q0 = *(const float4*)(p);
  const float4 q1 = *(const float4*)(p + 4);
  const float4 q2 = *(const float4*)(p + 16);
  const float4 q3 = *(const float4*)(p + 20);
  v16h f;
  f[0]  = (_Float16)q0.x; f[1]  = (_Float16)q0.y;
  f[2]  = (_Float16)q0.z; f[3]  = (_Float16)q0.w;
  f[4]  = (_Float16)q1.x; f[5]  = (_Float16)q1.y;
  f[6]  = (_Float16)q1.z; f[7]  = (_Float16)q1.w;
  f[8]  = (_Float16)q2.x; f[9]  = (_Float16)q2.y;
  f[10] = (_Float16)q2.z; f[11] = (_Float16)q2.w;
  f[12] = (_Float16)q3.x; f[13] = (_Float16)q3.y;
  f[14] = (_Float16)q3.z; f[15] = (_Float16)q3.w;
  return f;
}

// B fragment (32x16) from transposed (n-major, k-contiguous) f16 LDS.
__device__ __forceinline__ v16h load_bt_frag(const _Float16* bt, int ldb, int k0) {
  const int lane = threadIdx.x & 31;
  const int col  = lane & 15;
  const int kh   = lane >> 4;
  const _Float16* p = bt + col * ldb + k0 + kh * 8;
  v8h lo = *(const v8h*)(p);
  v8h hi = *(const v8h*)(p + 16);
  return frag_cat(lo, hi);
}

__device__ __forceinline__ float wave_sum(float v) {
  v += __shfl_xor(v, 16, 32);
  v += __shfl_xor(v, 8, 32);
  v += __shfl_xor(v, 4, 32);
  v += __shfl_xor(v, 2, 32);
  v += __shfl_xor(v, 1, 32);
  return v;
}

__device__ __forceinline__ float block_sum128(float v, volatile float* tmp4) {
  v = wave_sum(v);
  __syncthreads();
  if ((threadIdx.x & 31) == 0) tmp4[threadIdx.x >> 5] = v;
  __syncthreads();
  float r = tmp4[0] + tmp4[1] + tmp4[2] + tmp4[3];
  __syncthreads();
  return r;
}

// ---------------------------------------------------------------------------
// K1: LayerNorm(x) -> h.  One wave per token, 8 tokens per 256-thread block.
// ---------------------------------------------------------------------------
__global__ void __launch_bounds__(256) k_ln1(const float* __restrict__ x,
                                             const float* __restrict__ g,
                                             const float* __restrict__ b,
                                             float* __restrict__ h) {
  const int wave = threadIdx.x >> 5;
  const int lane = threadIdx.x & 31;
  const int tok  = blockIdx.x * 8 + wave;
  const float4 v = *(const float4*)(x + (size_t)tok * HIDn + lane * 4);
  float s = wave_sum(v.x + v.y + v.z + v.w);
  const float mu = s * (1.0f / 128.0f);
  const float dx = v.x - mu, dy = v.y - mu, dz = v.z - mu, dw = v.w - mu;
  const float var = wave_sum(dx * dx + dy * dy + dz * dz + dw * dw) * (1.0f / 128.0f);
  const float inv = rsqrtf(var + EPSc);
  const float4 gg = *(const float4*)(g + lane * 4);
  const float4 bb = *(const float4*)(b + lane * 4);
  float4 o;
  o.x = dx * inv * gg.x + bb.x;
  o.y = dy * inv * gg.y + bb.y;
  o.z = dz * inv * gg.z + bb.z;
  o.w = dw * inv * gg.w + bb.w;
  *(float4*)(h + (size_t)tok * HIDn + lane * 4) = o;
}

// ---------------------------------------------------------------------------
// K2: fused projection GEMM  Y[1024,512] = h @ [Wq|Wk|Wv|Ws] (+bias)
// Block = 128 thr (4 waves), each wave one 16x16 tile via 4 WMMAs (K=128).
// ---------------------------------------------------------------------------
__global__ void __launch_bounds__(128) k_proj(
    const float* __restrict__ h,
    const float* __restrict__ Wq, const float* __restrict__ bq,
    const float* __restrict__ Wk,
    const float* __restrict__ Wv, const float* __restrict__ bv,
    const float* __restrict__ Ws, const float* __restrict__ bs,
    float* __restrict__ q, float* __restrict__ kb,
    float* __restrict__ vb, float* __restrict__ sp) {
  __shared__ __align__(16) _Float16 aT[16 * 136];
  __shared__ __align__(16) _Float16 bT[64 * 136];

  const int m0   = blockIdx.x * 16;
  const int nblk = blockIdx.y * 64;
  const int sel  = nblk >> 7;     // 0..3 -> Wq, Wk, Wv, Ws
  const int col0 = nblk & 127;

  const float* W;
  const float* bias;
  float* O;
  switch (sel) {
    case 0: W = Wq; bias = bq;      O = q;  break;
    case 1: W = Wk; bias = nullptr; O = kb; break;
    case 2: W = Wv; bias = bv;      O = vb; break;
    default: W = Ws; bias = bs;     O = sp; break;
  }

  const int t = threadIdx.x;
  #pragma unroll
  for (int rep = 0; rep < 4; ++rep) {
    const int idx = rep * 512 + t * 4;
    const int row = idx >> 7, c = idx & 127;
    const float4 v = *(const float4*)(h + (size_t)(m0 + row) * HIDn + c);
    _Float16* d = aT + row * 136 + c;
    d[0] = (_Float16)v.x; d[1] = (_Float16)v.y;
    d[2] = (_Float16)v.z; d[3] = (_Float16)v.w;
  }
  for (int idx = t; idx < 8192; idx += 128) {
    const int k = idx >> 6, n = idx & 63;
    bT[n * 136 + k] = (_Float16)W[k * HDn + col0 + n];
  }
  __syncthreads();

  const int w = t >> 5;
  const _Float16* btw = bT + (w * 16) * 136;
  v8f acc = {};
  #pragma unroll
  for (int s = 0; s < 4; ++s) {
    v16h a  = load_a_frag(aT, 136, s * 32);
    v16h bb = load_bt_frag(btw, 136, s * 32);
    acc = __builtin_amdgcn_wmma_f32_16x16x32_f16(false, a, false, bb,
                                                 (short)0, acc, false, false);
  }
  const int lane  = t & 31;
  const int colm  = col0 + w * 16 + (lane & 15);
  const int rbase = m0 + ((lane >> 4) * 8);
  const float badd = bias ? bias[colm] : 0.0f;
  #pragma unroll
  for (int g8 = 0; g8 < 8; ++g8)
    O[(size_t)(rbase + g8) * HDn + colm] = acc[g8] + badd;
}

// ---------------------------------------------------------------------------
// K3: per-token mega-kernel.  One block (4 waves) per query token.
// r[b,i,:,:] streamed once via double-buffered GLOBAL_LOAD_ASYNC_TO_LDS_B128,
// projected with register-resident Wkr/Wvr WMMA fragments, flash softmax,
// then gate -> Wo residual -> LN2 -> FFN -> output.
// ---------------------------------------------------------------------------
__global__ void __launch_bounds__(128) k_attn(
    const float* __restrict__ r, const unsigned char* __restrict__ mask,
    const float* __restrict__ q, const float* __restrict__ kb,
    const float* __restrict__ vb, const float* __restrict__ sp,
    const float* __restrict__ hbuf, const float* __restrict__ x,
    const float* __restrict__ Wkr, const float* __restrict__ Wvr,
    const float* __restrict__ bvr,
    const float* __restrict__ Wg, const float* __restrict__ bg,
    const float* __restrict__ Wo, const float* __restrict__ bo,
    const float* __restrict__ ln2g, const float* __restrict__ ln2b,
    const float* __restrict__ W1, const float* __restrict__ b1,
    const float* __restrict__ W2, const float* __restrict__ b2,
    float* __restrict__ out) {
  __shared__ __align__(16) float rA32[2][16 * 132];  // async r tiles (f32)
  __shared__ __align__(16) float kvK[16 * 128];      // kb tile
  __shared__ __align__(16) float kvV[16 * 128];      // vb tile
  __shared__ __align__(16) float kF[16 * 128];       // k = kb + r@Wkr
  __shared__ __align__(16) float vF[16 * 128];       // v = vb + r@Wvr + bvr
  __shared__ float simT[16 * 8];
  __shared__ float pT[16 * 8];
  __shared__ float qv[128], hV[128];
  __shared__ float mh[8], sh[8], scl[8];
  __shared__ float accO[128];
  __shared__ float vecA[128], vecB[128], x1V[128];
  __shared__ float ff1V[512];
  __shared__ float tmp4[4];

  const int tok  = blockIdx.x;        // b*512 + i
  const int bI   = tok >> 9;
  const int t    = threadIdx.x;
  const int lane = t & 31;
  const int w    = t >> 5;

  qv[t]   = q[(size_t)tok * HDn + t];
  hV[t]   = hbuf[(size_t)tok * HIDn + t];
  accO[t] = 0.0f;
  if (t < 8) { mh[t] = -3.0e38f; sh[t] = 0.0f; }

  // B fragments for this wave's projection, resident in VGPRs all loop long.
  // waves 0,1 -> Wkr cols [0:64),[64:128); waves 2,3 -> Wvr same.
  const int    selv  = w >> 1;
  const float* Wsrc  = selv ? Wvr : Wkr;
  const int    nhalf = (w & 1) * 64;
  v16h bfrag[4][4];
  #pragma unroll
  for (int tt = 0; tt < 4; ++tt) {
    const int col = nhalf + tt * 16 + (lane & 15);
    const int khb = (lane >> 4) * 8;
    #pragma unroll
    for (int s = 0; s < 4; ++s) {
      const int k0 = s * 32 + khb;
      v16h f;
      #pragma unroll
      for (int e = 0; e < 16; ++e)
        f[e] = (_Float16)Wsrc[(size_t)(k0 + (e < 8 ? e : e + 8)) * HDn + col];
      bfrag[tt][s] = f;
    }
  }

  const float* rrow = r + (size_t)tok * Ln * HIDn;
  const float* kbB  = kb + (size_t)bI * Ln * HDn;
  const float* vbB  = vb + (size_t)bI * Ln * HDn;
  const unsigned char* mrow = mask + (size_t)tok * Ln;

  // ---- issue async copy of the first r tile (4 x b128 per thread) ----
  {
    const float* gR = rrow;            // contiguous 16x128 fp32 slab
    #pragma unroll
    for (int rep = 0; rep < 4; ++rep) {
      const int idx = rep * 128 + t;   // 4-float chunk id
      const int row = idx >> 5, c4 = idx & 31;
      async_ld_b128(lds_addr(&rA32[0][row * 132 + c4 * 4]), gR + idx * 4);
    }
  }

  for (int jt = 0; jt < 32; ++jt) {
    const int j0  = jt * 16;
    const int cur = jt & 1;
    // ---- stage kb/vb tiles (L2-hot, synchronous float4) ----
    #pragma unroll
    for (int rep = 0; rep < 4; ++rep) {
      const int idx = rep * 512 + t * 4;
      *(float4*)(kvK + idx) = *(const float4*)(kbB + (size_t)j0 * HDn + idx);
      *(float4*)(kvV + idx) = *(const float4*)(vbB + (size_t)j0 * HDn + idx);
    }
    // ---- issue async copy of NEXT r tile into the other buffer ----
    if (jt + 1 < 32) {
      const float* gR = rrow + (size_t)(j0 + 16) * HIDn;
      #pragma unroll
      for (int rep = 0; rep < 4; ++rep) {
        const int idx = rep * 128 + t;
        const int row = idx >> 5, c4 = idx & 31;
        async_ld_b128(lds_addr(&rA32[cur ^ 1][row * 132 + c4 * 4]), gR + idx * 4);
      }
      wait_async_le4();   // retire current tile's batch; next stays in flight
    } else {
      wait_async_le0();
    }
    __syncthreads();

    // ---- WMMA: this wave's half of rk or rv, then add base (+bvr) ----
    {
      const float* rCur = rA32[cur];
      float*       dstF = selv ? vF : kF;
      const float* addT = selv ? kvV : kvK;
      #pragma unroll
      for (int tt = 0; tt < 4; ++tt) {
        v8f acc = {};
        #pragma unroll
        for (int s = 0; s < 4; ++s) {
          v16h a = load_a_frag_f32(rCur, 132, s * 32);
          acc = __builtin_amdgcn_wmma_f32_16x16x32_f16(false, a, false,
                                                       bfrag[tt][s], (short)0,
                                                       acc, false, false);
        }
        const int colm = nhalf + tt * 16 + (lane & 15);
        const int rb   = (lane >> 4) * 8;
        const float ba = selv ? bvr[colm] : 0.0f;
        #pragma unroll
        for (int g8 = 0; g8 < 8; ++g8) {
          const int jl = rb + g8;
          dstF[jl * 128 + colm] = acc[g8] + addT[jl * 128 + colm] + ba;
        }
      }
    }
    __syncthreads();

    // ---- sim[j,h] = 0.25 * <q_h, k_j,h>, masked ----
    {
      const int jl = t >> 3, hh = t & 7;
      float d = 0.0f;
      #pragma unroll
      for (int dd = 0; dd < 16; ++dd)
        d += qv[hh * 16 + dd] * kF[jl * 128 + hh * 16 + dd];
      d *= 0.25f;
      if (!mrow[j0 + jl]) d = -100.0f;
      simT[jl * 8 + hh] = d;
    }
    __syncthreads();

    // ---- online softmax state update (one thread per head) ----
    if (t < 8) {
      const float mold = mh[t];
      float tm = mold;
      #pragma unroll
      for (int jl = 0; jl < 16; ++jl) tm = fmaxf(tm, simT[jl * 8 + t]);
      const float sc = __expf(mold - tm);
      float ss = 0.0f;
      #pragma unroll
      for (int jl = 0; jl < 16; ++jl) {
        const float p = __expf(simT[jl * 8 + t] - tm);
        pT[jl * 8 + t] = p;
        ss += p;
      }
      sh[t] = sh[t] * sc + ss;
      mh[t] = tm;
      scl[t] = sc;
    }
    __syncthreads();

    // ---- rescale + accumulate weighted v ----
    {
      const int hh = t >> 4, dd = t & 15;
      float a = accO[t] * scl[hh];
      #pragma unroll
      for (int jl = 0; jl < 16; ++jl)
        a += pT[jl * 8 + hh] * vF[jl * 128 + hh * 16 + dd];
      accO[t] = a;
    }
    __syncthreads();
  }

  // ---- finalize attention output ("inputs") ----
  const float inp = accO[t] / sh[t >> 4];
  vecA[t] = inp;
  __syncthreads();

  // ---- gate: g = sigmoid([inputs, h] @ Wg + bg); agg = lerp ----
  {
    float gd = bg[t];
    #pragma unroll 4
    for (int m = 0; m < 128; ++m) gd += vecA[m] * Wg[m * 128 + t];
    #pragma unroll 4
    for (int m = 0; m < 128; ++m) gd += hV[m] * Wg[(128 + m) * 128 + t];
    const float gs = 1.0f / (1.0f + __expf(-gd));
    const float sv = sp[(size_t)tok * HDn + t];
    vecB[t] = inp + gs * (sv - inp);
  }
  __syncthreads();

  // ---- x1 = x + agg @ Wo + bo ----
  {
    float v = x[(size_t)tok * HIDn + t] + bo[t];
    #pragma unroll 4
    for (int m = 0; m < 128; ++m) v += vecB[m] * Wo[m * 128 + t];
    x1V[t] = v;
  }
  __syncthreads();

  // ---- LN2 ----
  const float mu  = block_sum128(x1V[t], tmp4) * (1.0f / 128.0f);
  const float dxv = x1V[t] - mu;
  const float var = block_sum128(dxv * dxv, tmp4) * (1.0f / 128.0f);
  vecA[t] = dxv * rsqrtf(var + EPSc) * ln2g[t] + ln2b[t];
  __syncthreads();

  // ---- FFN: ff1 = relu(x2 @ W1 + b1)  (512 wide) ----
  #pragma unroll
  for (int rep = 0; rep < 4; ++rep) {
    const int k = rep * 128 + t;
    float v = b1[k];
    for (int n = 0; n < 128; ++n) v += vecA[n] * W1[n * 512 + k];
    ff1V[k] = fmaxf(v, 0.0f);
  }
  __syncthreads();

  // ---- out = x1 + ff1 @ W2 + b2 ----
  {
    float v = x1V[t] + b2[t];
    for (int kk = 0; kk < 512; ++kk) v += ff1V[kk] * W2[kk * 128 + t];
    out[(size_t)tok * HIDn + t] = v;
  }
}

// ---------------------------------------------------------------------------
// Host launcher
// ---------------------------------------------------------------------------
extern "C" void kernel_launch(void* const* d_in, const int* in_sizes, int n_in,
                              void* d_out, int out_size, void* d_ws, size_t ws_size,
                              hipStream_t stream) {
  (void)in_sizes; (void)n_in; (void)out_size; (void)ws_size;
  const float* x    = (const float*)d_in[0];
  const float* r    = (const float*)d_in[1];
  const unsigned char* mask = (const unsigned char*)d_in[2];
  const float* ln1g = (const float*)d_in[3];
  const float* ln1b = (const float*)d_in[4];
  const float* Wq   = (const float*)d_in[5];
  const float* bq   = (const float*)d_in[6];
  const float* Wk   = (const float*)d_in[7];
  const float* Wv   = (const float*)d_in[8];
  const float* bv   = (const float*)d_in[9];
  const float* Wkr  = (const float*)d_in[10];
  const float* Wvr  = (const float*)d_in[11];
  const float* bvr  = (const float*)d_in[12];
  const float* Ws   = (const float*)d_in[13];
  const float* bs   = (const float*)d_in[14];
  const float* Wg   = (const float*)d_in[15];
  const float* bg   = (const float*)d_in[16];
  const float* Wo   = (const float*)d_in[17];
  const float* bo   = (const float*)d_in[18];
  const float* ln2g = (const float*)d_in[19];
  const float* ln2b = (const float*)d_in[20];
  const float* W1   = (const float*)d_in[21];
  const float* b1   = (const float*)d_in[22];
  const float* W2   = (const float*)d_in[23];
  const float* b2   = (const float*)d_in[24];

  float* ws = (float*)d_ws;
  float* h  = ws;                    // 1024*128
  float* q  = ws + 131072;
  float* kb = ws + 262144;
  float* vb = ws + 393216;
  float* sp = ws + 524288;           // total 655360 floats (2.5 MB)

  k_ln1<<<NTOK / 8, 256, 0, stream>>>(x, ln1g, ln1b, h);
  k_proj<<<dim3(NTOK / 16, 8), 128, 0, stream>>>(h, Wq, bq, Wk, Wv, bv, Ws, bs,
                                                 q, kb, vb, sp);
  k_attn<<<NTOK, 128, 0, stream>>>(r, mask, q, kb, vb, sp, h, x, Wkr, Wvr, bvr,
                                   Wg, bg, Wo, bo, ln2g, ln2b, W1, b1, W2, b2,
                                   (float*)d_out);
}
